// RPN_79989470920833
// MI455X (gfx1250) — matrix operand order repaired and checked by reading, not compiled, not run on hardware
//
#include <hip/hip_runtime.h>
#include <hip/hip_bf16.h>
#include <math.h>

typedef __attribute__((ext_vector_type(16))) _Float16 v16h;
typedef __attribute__((ext_vector_type(8)))  float    v8f;

#define IMG_H 360
#define IMG_W 1200
#define BEV_H 700
#define BEV_W 800
#define C_IN 32
#define N_ANCH 16384
#define NMS_K 1024
#define NMS_T 0.8f
#define X_MIN_C (-40.0f)
#define Z_MAX_C (70.0f)

// ------- 1x1 conv + relu + mask bake: float4 streaming (b128 loads) ---------
// Dominant traffic (~127MB); 4 pixels/thread -> global_load_b128, 512B per
// vmem instruction per wave, to sustain the 23.3 TB/s HBM roof.
__global__ void conv11_relu_mask_v4(const float4* __restrict__ in,
                                    const float* __restrict__ w,
                                    const float* __restrict__ b,
                                    const float* __restrict__ mask,
                                    float4* __restrict__ out, int HW4) {
  int p = blockIdx.x * blockDim.x + threadIdx.x;
  if (p >= HW4) return;
  float bb = b[0];
  float ax = bb, ay = bb, az = bb, aw = bb;
#pragma unroll
  for (int c = 0; c < C_IN; ++c) {
    float4 v = in[(size_t)c * HW4 + p];
    float wc = w[c];
    ax = fmaf(wc, v.x, ax);
    ay = fmaf(wc, v.y, ay);
    az = fmaf(wc, v.z, az);
    aw = fmaf(wc, v.w, aw);
  }
  float m = mask[0];
  float4 o;
  o.x = m * fmaxf(ax, 0.0f);
  o.y = m * fmaxf(ay, 0.0f);
  o.z = m * fmaxf(az, 0.0f);
  o.w = m * fmaxf(aw, 0.0f);
  out[p] = o;
}

// ---------------- ROI align (bilinear, L2-resident feature maps) ------------
__device__ __forceinline__ float bilinear_samp(const float* __restrict__ f, int H, int W,
                                               float y, float x) {
  if (!(y > -1.0f && y < (float)H && x > -1.0f && x < (float)W)) return 0.0f;
  float yc = fminf(fmaxf(y, 0.0f), (float)(H - 1));
  float xc = fminf(fmaxf(x, 0.0f), (float)(W - 1));
  int yl = (int)floorf(yc), xl = (int)floorf(xc);
  int yh = min(yl + 1, H - 1), xh = min(xl + 1, W - 1);
  float ly = yc - (float)yl, lx = xc - (float)xl;
  float v00 = f[(size_t)yl * W + xl], v01 = f[(size_t)yl * W + xh];
  float v10 = f[(size_t)yh * W + xl], v11 = f[(size_t)yh * W + xh];
  return (1.0f - ly) * (1.0f - lx) * v00 + (1.0f - ly) * lx * v01 +
         ly * (1.0f - lx) * v10 + ly * lx * v11;
}

__device__ __forceinline__ float roi_bin(const float* __restrict__ f, int H, int W,
                                         const float* __restrict__ roi, int oy, int ox) {
  float x0 = roi[1] - 0.5f, y0 = roi[2] - 0.5f;
  float rw = roi[3] - roi[1], rh = roi[4] - roi[2];
  float acc = 0.0f;
#pragma unroll
  for (int sy = 0; sy < 2; ++sy) {
    int gy = oy * 2 + sy;
    float fy = ((float)(gy >> 1) + ((float)(gy & 1) + 0.5f) * 0.5f) * (1.0f / 3.0f);
    float y = y0 + rh * fy;
#pragma unroll
    for (int sx = 0; sx < 2; ++sx) {
      int gx = ox * 2 + sx;
      float fx = ((float)(gx >> 1) + ((float)(gx & 1) + 0.5f) * 0.5f) * (1.0f / 3.0f);
      float x = x0 + rw * fx;
      acc += bilinear_samp(f, H, W, y, x);
    }
  }
  return acc * 0.25f;
}

__global__ void roi_fuse(const float* __restrict__ fi, const float* __restrict__ fb,
                         const float* __restrict__ ai, const float* __restrict__ ab,
                         const float* __restrict__ m1, const float* __restrict__ m2,
                         float* __restrict__ feats) {
  int t = blockIdx.x * blockDim.x + threadIdx.x;
  if (t >= N_ANCH * 9) return;
  int roi = t / 9, bin = t % 9;
  int oy = bin / 3, ox = bin % 3;
  float vi = roi_bin(fi, IMG_H, IMG_W, ai + (size_t)roi * 5, oy, ox);
  float vb = roi_bin(fb, BEV_H, BEV_W, ab + (size_t)roi * 5, oy, ox);
  feats[t] = (vi + vb) / (m1[0] + m2[0]);
}

// ---------------- FC head as WMMA: (16384x9) @ (9x8), K padded to 32 --------
// A(16x32 f16): lanes 0-15 hold M=lane, halves 0..8 = K0..8; lanes 16-31 hold
// zeros (K padding). B(32x16 f16): lanes 0-15
// hold N=lane, half e = K=e. D(16x16 f32) per ISA C/D layout.
__global__ void fc_wmma(const float* __restrict__ feats,
                        const float* __restrict__ obj_w, const float* __restrict__ obj_b,
                        const float* __restrict__ off_w, const float* __restrict__ off_b,
                        float* __restrict__ fc_out) {
  int lane = threadIdx.x & 31;
  int tile = blockIdx.x * (blockDim.x >> 5) + (threadIdx.x >> 5);  // 0..1023

  v16h a = {};
  int m = lane & 15;
  const float* frow = feats + (size_t)(tile * 16 + m) * 9;
  if (lane < 16) {
#pragma unroll
    for (int e = 0; e < 8; ++e) a[e] = (_Float16)frow[e];
    a[8] = (_Float16)frow[8];
  }

  v16h bm = {};
  if (lane < 8) {
    int n = lane;  // output column: 0..1 = objectness, 2..7 = offsets
#pragma unroll
    for (int k = 0; k < 9; ++k) {
      float wv = (n < 2) ? obj_w[k * 2 + n] : off_w[k * 6 + (n - 2)];
      bm[k] = (_Float16)wv;
    }
  }

  v8f c = {};
  c = __builtin_amdgcn_wmma_f32_16x16x32_f16(false, a, false, bm, (short)0, c, false, false);

  int n = lane & 15;
  int hi = lane >> 4;
  if (n < 8) {
    float bias = (n < 2) ? obj_b[n] : off_b[n - 2];
#pragma unroll
    for (int e = 0; e < 8; ++e) {
      int mm = e + hi * 8;
      fc_out[(size_t)(tile * 16 + mm) * 8 + n] = c[e] + bias;
    }
  }
}

// ---------------- box decode + sigmoid score --------------------------------
__global__ void decode_boxes(const float* __restrict__ fc, const float* __restrict__ fa,
                             float* __restrict__ reg, float* __restrict__ boxes,
                             float* __restrict__ scores) {
  int i = blockIdx.x * blockDim.x + threadIdx.x;
  if (i >= N_ANCH) return;
  const float* o = fc + (size_t)i * 8;
  const float* a = fa + (size_t)i * 6;
  float r[6];
#pragma unroll
  for (int j = 0; j < 3; ++j) r[j] = o[2 + j] * a[3 + j] + a[j];
#pragma unroll
  for (int j = 0; j < 3; ++j) r[3 + j] = __expf(o[5 + j]) * a[3 + j];
#pragma unroll
  for (int j = 0; j < 6; ++j) reg[(size_t)i * 6 + j] = r[j];
  float bx1 = r[0] - r[3] * 0.5f - X_MIN_C;
  float bx2 = r[0] + r[3] * 0.5f - X_MIN_C;
  float bz1 = Z_MAX_C - (r[2] + r[5] * 0.5f);
  float bz2 = Z_MAX_C - (r[2] - r[5] * 0.5f);
  boxes[(size_t)i * 4 + 0] = bx1;
  boxes[(size_t)i * 4 + 1] = bz1;
  boxes[(size_t)i * 4 + 2] = bx2;
  boxes[(size_t)i * 4 + 3] = bz2;
  scores[i] = 1.0f / (1.0f + __expf(o[0] - o[1]));  // softmax column 1
}

// ---------------- sequential NMS: 1 workgroup (32 wave32s) ------------------
// Scores live in registers (each thread exclusively owns slots t+k*1024).
// Argmax = shfl_xor wave reduce (5 hops) -> 32 LDS partials -> one-wave
// shfl reduce. 3 barriers per iteration instead of a 1024-wide LDS tree.
// Boxes (256KB) stay L2-resident; areas recomputed bit-exactly from coords.
__global__ void __launch_bounds__(1024) nms_kernel(const float4* __restrict__ boxes,
                                                   const float* __restrict__ scores_in,
                                                   int* __restrict__ top_idx) {
  __shared__ float s_val[32];
  __shared__ int s_id[32];
  __shared__ float s_box[5];
  __shared__ int s_j;
  int t = threadIdx.x;

  float sc[16];
#pragma unroll
  for (int k = 0; k < 16; ++k) sc[k] = scores_in[t + k * 1024];

  for (int it = 0; it < NMS_K; ++it) {
    // local scan (ascending index => strict '>' keeps first max)
    float bv = sc[0];
    int bi = t;
#pragma unroll
    for (int k = 1; k < 16; ++k) {
      if (sc[k] > bv) { bv = sc[k]; bi = t + k * 1024; }
    }
    // wave32 shuffle reduction, min-index tiebreak (jnp.argmax semantics)
#pragma unroll
    for (int d = 16; d > 0; d >>= 1) {
      float ov = __shfl_xor(bv, d, 32);
      int oi = __shfl_xor(bi, d, 32);
      if (ov > bv || (ov == bv && oi < bi)) { bv = ov; bi = oi; }
    }
    if ((t & 31) == 0) { s_val[t >> 5] = bv; s_id[t >> 5] = bi; }
    __syncthreads();
    if (t < 32) {
      float v = s_val[t];
      int idx = s_id[t];
#pragma unroll
      for (int d = 16; d > 0; d >>= 1) {
        float ov = __shfl_xor(v, d, 32);
        int oi = __shfl_xor(idx, d, 32);
        if (ov > v || (ov == v && oi < idx)) { v = ov; idx = oi; }
      }
      if (t == 0) {
        s_j = idx;
        top_idx[it] = idx;
        float4 bj = boxes[idx];
        s_box[0] = bj.x;
        s_box[1] = bj.y;
        s_box[2] = bj.z;
        s_box[3] = bj.w;
        s_box[4] = (bj.z - bj.x) * (bj.w - bj.y);
      }
    }
    __syncthreads();
    int j = s_j;
    float jx1 = s_box[0], jz1 = s_box[1], jx2 = s_box[2], jz2 = s_box[3], ja = s_box[4];
#pragma unroll
    for (int k = 0; k < 16; ++k) {
      int idx = t + k * 1024;
      float4 bk = boxes[idx];
      float ak = (bk.z - bk.x) * (bk.w - bk.y);
      float iw = fmaxf(fminf(jx2, bk.z) - fmaxf(jx1, bk.x), 0.0f);
      float ih = fmaxf(fminf(jz2, bk.w) - fmaxf(jz1, bk.y), 0.0f);
      float inter = iw * ih;
      float iou = inter / (ja + ak - inter + 1e-8f);
      if (idx == j || iou > NMS_T) sc[k] = -__builtin_inff();
    }
    __syncthreads();
  }
}

// ---------------- gather outputs: reg_anchors[top] (1024x6) ++ obj[top] (1024x2)
__global__ void gather_out(const int* __restrict__ top_idx, const float* __restrict__ reg,
                           const float* __restrict__ fc, float* __restrict__ out) {
  int i = blockIdx.x * blockDim.x + threadIdx.x;
  if (i >= NMS_K) return;
  int j = top_idx[i];
#pragma unroll
  for (int k = 0; k < 6; ++k) out[(size_t)i * 6 + k] = reg[(size_t)j * 6 + k];
  out[(size_t)NMS_K * 6 + i * 2 + 0] = fc[(size_t)j * 8 + 0];
  out[(size_t)NMS_K * 6 + i * 2 + 1] = fc[(size_t)j * 8 + 1];
}

extern "C" void kernel_launch(void* const* d_in, const int* in_sizes, int n_in,
                              void* d_out, int out_size, void* d_ws, size_t ws_size,
                              hipStream_t stream) {
  const float* img_map    = (const float*)d_in[0];
  const float* bev_map    = (const float*)d_in[1];
  const float* anchors_i  = (const float*)d_in[2];
  const float* anchors_b  = (const float*)d_in[3];
  const float* filtered   = (const float*)d_in[4];
  const float* img_mask   = (const float*)d_in[5];
  const float* bev_mask   = (const float*)d_in[6];
  const float* conv_img_w = (const float*)d_in[7];
  const float* conv_img_b = (const float*)d_in[8];
  const float* conv_bev_w = (const float*)d_in[9];
  const float* conv_bev_b = (const float*)d_in[10];
  const float* obj_w      = (const float*)d_in[11];
  const float* obj_b      = (const float*)d_in[12];
  const float* off_w      = (const float*)d_in[13];
  const float* off_b      = (const float*)d_in[14];

  float* ws = (float*)d_ws;
  size_t off = 0;
  float* feat_img = ws + off; off += (size_t)IMG_H * IMG_W;   // 432000
  float* feat_bev = ws + off; off += (size_t)BEV_H * BEV_W;   // 560000
  float* feats    = ws + off; off += (size_t)N_ANCH * 9;      // 147456
  float* fc_out   = ws + off; off += (size_t)N_ANCH * 8;      // 131072
  float* reg      = ws + off; off += (size_t)N_ANCH * 6;      // 98304
  float* boxes    = ws + off; off += (size_t)N_ANCH * 4;      // 65536
  float* scores   = ws + off; off += (size_t)N_ANCH;
  int*   top_idx  = (int*)(ws + off); off += NMS_K;

  int hw4I = (IMG_H * IMG_W) / 4, hw4B = (BEV_H * BEV_W) / 4;
  conv11_relu_mask_v4<<<(hw4I + 255) / 256, 256, 0, stream>>>(
      (const float4*)img_map, conv_img_w, conv_img_b, img_mask, (float4*)feat_img, hw4I);
  conv11_relu_mask_v4<<<(hw4B + 255) / 256, 256, 0, stream>>>(
      (const float4*)bev_map, conv_bev_w, conv_bev_b, bev_mask, (float4*)feat_bev, hw4B);
  roi_fuse<<<(N_ANCH * 9 + 255) / 256, 256, 0, stream>>>(feat_img, feat_bev, anchors_i,
                                                         anchors_b, img_mask, bev_mask, feats);
  // 1024 tiles of 16 ROIs, 8 waves per 256-thread block -> 128 blocks
  fc_wmma<<<128, 256, 0, stream>>>(feats, obj_w, obj_b, off_w, off_b, fc_out);
  decode_boxes<<<(N_ANCH + 255) / 256, 256, 0, stream>>>(fc_out, filtered, reg, boxes, scores);
  nms_kernel<<<1, 1024, 0, stream>>>((const float4*)boxes, scores, top_idx);
  gather_out<<<(NMS_K + 255) / 256, 256, 0, stream>>>(top_idx, reg, fc_out, (float*)d_out);
}
